// BidirectionalJointUpdater_17978733101719
// MI455X (gfx1250) — compile-verified, matrix operand order checked by reading, not run on hardware
//
#include <hip/hip_runtime.h>
#include <hip/hip_bf16.h>
#include <stdint.h>

typedef unsigned short u16;
typedef __attribute__((ext_vector_type(16))) __bf16 v16bf;
typedef __attribute__((ext_vector_type(8)))  float  v8f;

// ---- TDM availability (this toolchain: 6-arg clang-23 form) ----------------
#if defined(__has_builtin)
#if __has_builtin(__builtin_amdgcn_tensor_load_to_lds) && \
    __has_builtin(__builtin_amdgcn_s_wait_tensorcnt)
#define USE_TDM 1
#endif
#endif
#ifndef USE_TDM
#define USE_TDM 0
#endif

typedef unsigned int u32x4 __attribute__((ext_vector_type(4)));
typedef int          i32x8 __attribute__((ext_vector_type(8)));
typedef int          i32x4 __attribute__((ext_vector_type(4)));

// ---------------------------------------------------------------------------
// helpers
// ---------------------------------------------------------------------------
__device__ __forceinline__ u16 f2bf(float f) {
    unsigned u = __float_as_uint(f);
    u += 0x7FFFu + ((u >> 16) & 1u);         // round-to-nearest-even
    return (u16)(u >> 16);
}

union Frag16 { uint4 q[2]; v16bf v; };

// 16-lane-major WMMA fragment (A 16x32 / B 32x16): lead index (M or N) = lane&15,
// K contiguous. Lanes 0-15 hold K {0..7,16..23}; lanes 16-31 hold K {8..15,24..31}.
__device__ __forceinline__ v16bf load_frag16(const u16* base, int lead, int ldm,
                                             int kbase, int lane) {
    const int h = lane >> 4;
    const int r = lane & 15;
    const u16* p = base + (size_t)(lead + r) * (size_t)ldm + kbase + h * 8;
    Frag16 f;
    f.q[0] = *(const uint4*)(p);
    f.q[1] = *(const uint4*)(p + 16);
    return f.v;
}

__device__ __forceinline__ v8f wmma_bf16(v16bf a, v16bf b, v8f c) {
    return __builtin_amdgcn_wmma_f32_16x16x32_bf16(false, a, false, b,
                                                   (short)0, c, false, false);
}

#if USE_TDM
// Low 32 bits of a generic LDS pointer are the wave-relative LDS byte address.
__device__ __forceinline__ unsigned lds_addr_of(const void* p) {
    return (unsigned)(unsigned long long)p;
}

// Issue one TENSOR_LOAD_TO_LDS for a 2D tile of 2-byte elements.
//  tile_d0 : elements per line, tile_d1 : number of lines (0 => single line)
//  stride0 : elements between consecutive lines in global memory
// D# bit layout per CDNA5 ISA.
__device__ __forceinline__ void tdm_load_2d(const void* gsrc, unsigned lds_addr,
                                            unsigned tile_d0, unsigned tile_d1,
                                            unsigned long long tensor_d0,
                                            unsigned tensor_d1,
                                            unsigned long long stride0) {
    unsigned long long ga = (unsigned long long)gsrc;
    u32x4 g0;
    g0[0] = 1u;                                           // count=1, user descriptor
    g0[1] = lds_addr;                                     // lds_addr[31:0]
    g0[2] = (unsigned)(ga & 0xFFFFFFFFu);                 // global_addr[31:0]
    g0[3] = (unsigned)((ga >> 32) & 0x01FFFFFFu)          // global_addr[56:32]
          | (2u << 30);                                   // type = 2 ("image")
    i32x8 g1;
    g1[0] = (int)(1u << 16);                              // data_size=1 (2 bytes)
    g1[1] = (int)((unsigned)(tensor_d0 & 0xFFFFu) << 16); // tensor_dim0[15:0]
    g1[2] = (int)(((unsigned)(tensor_d0 >> 16) & 0xFFFFu)
          | ((tensor_d1 & 0xFFFFu) << 16));               // tensor_dim0 hi | tensor_dim1 lo
    g1[3] = (int)(((tensor_d1 >> 16) & 0xFFFFu)
          | ((tile_d0 & 0xFFFFu) << 16));                 // tensor_dim1 hi | tile_dim0
    g1[4] = (int)(tile_d1 & 0xFFFFu);                     // tile_dim1 | tile_dim2(=0)
    g1[5] = (int)(stride0 & 0xFFFFFFFFu);                 // tensor_dim0_stride[31:0]
    g1[6] = (int)((stride0 >> 32) & 0xFFFFu);             // stride0 hi | stride1 lo(=0)
    g1[7] = 0;                                            // stride1 hi(=0)
    i32x4 z4 = {0, 0, 0, 0};                              // groups 2/3 unused (<=2D)
    i32x8 z8 = {0, 0, 0, 0, 0, 0, 0, 0};                  // 5th operand (unused here)
    __builtin_amdgcn_tensor_load_to_lds(g0, g1, z4, z4, z8, 0);
}
#endif

// ---------------------------------------------------------------------------
// Kernel 0: fp32 -> bf16, row-major + transposed copies
//   ws (u16 units): W[D*D] | user[N*D] | item[M*D] | userT[D*N] | itemT[D*M]
// ---------------------------------------------------------------------------
__global__ void convert_kernel(const float* __restrict__ user,
                               const float* __restrict__ item,
                               const float* __restrict__ W,
                               u16* __restrict__ user_bf, u16* __restrict__ item_bf,
                               u16* __restrict__ userT,   u16* __restrict__ itemT,
                               u16* __restrict__ w_bf,
                               int N, int M, int D) {
    size_t i = (size_t)blockIdx.x * blockDim.x + threadIdx.x;
    size_t nU = (size_t)N * D, nI = (size_t)M * D, nW = (size_t)D * D;
    if (i < nU) {
        int u = (int)(i / D), f = (int)(i % D);
        u16 b = f2bf(user[i]);
        user_bf[i] = b;
        userT[(size_t)f * N + u] = b;
    } else if (i < nU + nI) {
        size_t j = i - nU;
        int it = (int)(j / D), f = (int)(j % D);
        u16 b = f2bf(item[j]);
        item_bf[j] = b;
        itemT[(size_t)f * M + it] = b;
    } else if (i < nU + nI + nW) {
        size_t j = i - nU - nI;
        w_bf[j] = f2bf(W[j]);
    }
}

// ---------------------------------------------------------------------------
// Kernel 1: out = relu(x @ W^T + bias), 16 rows/wave, bf16 WMMA
// ---------------------------------------------------------------------------
__global__ void transform_kernel(const u16* __restrict__ x_bf,
                                 const u16* __restrict__ w_bf,
                                 const float* __restrict__ bias,
                                 float* __restrict__ out, int R, int D) {
    const int lane = threadIdx.x & 31;
    const int h    = lane >> 4;
    const int l15  = lane & 15;
    const int wave = blockIdx.x * (blockDim.x >> 5) + (threadIdx.x >> 5);
    const int rb   = wave * 16;

    v16bf a[4];
#pragma unroll
    for (int kc = 0; kc < 4; ++kc) a[kc] = load_frag16(x_bf, rb, D, kc * 32, lane);

#pragma unroll
    for (int nt = 0; nt < 8; ++nt) {
        v8f c = {};
        v16bf bw[4];
#pragma unroll
        for (int kc = 0; kc < 4; ++kc) bw[kc] = load_frag16(w_bf, nt * 16, D, kc * 32, lane);
#pragma unroll
        for (int kc = 0; kc < 4; ++kc) c = wmma_bf16(a[kc], bw[kc], c);

        const int o = nt * 16 + l15;
        const float bv = bias[o];
#pragma unroll
        for (int r = 0; r < 8; ++r) {
            float v = c[r] + bv;
            out[(size_t)(rb + r + 8 * h) * D + o] = v > 0.0f ? v : 0.0f;
        }
    }
}

// ---------------------------------------------------------------------------
// Kernel 2: fused masked-softmax attention, flash style, single pass.
// Block = 4 waves x 16 rows. The block walks all 32-column tiles; each tile
// (8 KB row-major + 8 KB transposed) is staged into double-buffered LDS once,
// by the Tensor Data Mover when available (one DMA pair per tile, TENSORcnt),
// else by a register-staging pipeline. Next tile is in flight during compute.
// ---------------------------------------------------------------------------
template <bool USER_SIDE>
__global__ void attn_kernel(const u16* __restrict__ rows_bf,   // [R x 128]
                            const u16* __restrict__ cols_bf,   // [C x 128]
                            const u16* __restrict__ colsT_bf,  // [128 x C]
                            const int* __restrict__ adj,       // [N x M]
                            float* __restrict__ out, int R, int C) {
    const int lane = threadIdx.x & 31;
    const int h    = lane >> 4;
    const int l15  = lane & 15;
    const int wid  = threadIdx.x >> 5;
    const int tid  = threadIdx.x;
    const int rb   = (blockIdx.x * 4 + wid) * 16;   // this wave's 16 rows

    __shared__ __align__(16) u16 sRM[2][32 * 128];  // [col][feature]
    __shared__ __align__(16) u16 sTR[2][128 * 32];  // [feature][col]
    __shared__ __align__(16) u16 sP[4][16 * 32];    // per-wave P staging
    u16* myP = sP[wid];

#if USE_TDM
    // One DMA pair per tile, issued by wave 0 only (TDM ignores EXEC and
    // issues once per executing wave; TENSORcnt is wave 0's to wait on).
    auto stage_issue = [&](int cb, int buf) {
        if (wid == 0) {
            // row-major tile: rows cb..cb+31 are one contiguous 8 KB line
            tdm_load_2d(cols_bf + (size_t)cb * 128, lds_addr_of(sRM[buf]),
                        /*tile_d0=*/32 * 128, /*tile_d1=*/0,
                        /*tensor_d0=*/32 * 128, /*tensor_d1=*/1,
                        /*stride0=*/32 * 128);
            // transposed tile: 128 lines x 32 elements, line stride C elements
            tdm_load_2d(colsT_bf + cb, lds_addr_of(sTR[buf]),
                        /*tile_d0=*/32, /*tile_d1=*/128,
                        /*tensor_d0=*/(unsigned long long)C, /*tensor_d1=*/128,
                        /*stride0=*/(unsigned long long)C);
        }
    };
    auto stage_wait = [&]() {
        if (wid == 0) __builtin_amdgcn_s_wait_tensorcnt(0);
        __syncthreads();
    };
#else
    uint4 rmReg[4], trReg[4];                        // 64 B/thread/buffer
    auto stage_load = [&](int cb) {
        const uint4* gRM = (const uint4*)(cols_bf + (size_t)cb * 128);
#pragma unroll
        for (int j = 0; j < 4; ++j) rmReg[j] = gRM[tid * 4 + j];
        const uint4* gTR = (const uint4*)(colsT_bf + (size_t)tid * C + cb);
#pragma unroll
        for (int j = 0; j < 4; ++j) trReg[j] = gTR[j];
    };
    auto stage_store = [&](int buf) {
        uint4* lRM = (uint4*)sRM[buf];
#pragma unroll
        for (int j = 0; j < 4; ++j) lRM[tid * 4 + j] = rmReg[j];
        uint4* lTR = (uint4*)sTR[buf];
#pragma unroll
        for (int j = 0; j < 4; ++j) lTR[tid * 4 + j] = trReg[j];
    };
#endif

    v16bf a[4];
#pragma unroll
    for (int kc = 0; kc < 4; ++kc) a[kc] = load_frag16(rows_bf, rb, 128, kc * 32, lane);

    v8f ctx[8];
#pragma unroll
    for (int nt = 0; nt < 8; ++nt) ctx[nt] = (v8f){};
    float dsum[8] = {0, 0, 0, 0, 0, 0, 0, 0};

#if USE_TDM
    stage_issue(0, 0);
    stage_wait();
#else
    stage_load(0);
    stage_store(0);
    __syncthreads();
#endif

    int bufc = 0;
    for (int cb = 0; cb < C; cb += 32, bufc ^= 1) {
        const bool more = (cb + 32) < C;
        if (more) {
#if USE_TDM
            stage_issue(cb + 32, bufc ^ 1);   // DMA next tile during compute
#else
            stage_load(cb + 32);
#endif
            // speculative prefetch of next tile's mask lines (the HBM stream)
            const size_t pidx = USER_SIDE
                ? (size_t)(rb + l15) * (size_t)C + (cb + 32 + h * 16)
                : (size_t)(cb + 32 + lane) * (size_t)R + rb;
            __builtin_prefetch(adj + pidx, 0, 1);
        }

        // --- S = rows @ colsT : two 16x16 sub-tiles from the LDS tile ---
#pragma unroll
        for (int t = 0; t < 2; ++t) {
            v16bf b1[4];
#pragma unroll
            for (int kc = 0; kc < 4; ++kc)
                b1[kc] = load_frag16(sRM[bufc], t * 16, 128, kc * 32, lane);
            v8f s = {};
#pragma unroll
            for (int kc = 0; kc < 4; ++kc) s = wmma_bf16(a[kc], b1[kc], s);

            const int col = cb + t * 16 + l15;
#pragma unroll
            for (int r = 0; r < 8; ++r) {
                const int row = rb + r + 8 * h;
                const size_t midx = USER_SIDE ? (size_t)row * (size_t)C + col
                                              : (size_t)col * (size_t)R + row;
                const int mv = __builtin_nontemporal_load(adj + midx);
                float p = (mv > 0) ? __expf(s[r]) : 0.0f;   // exp(-1e9) == 0
                dsum[r] += p;
                myP[(r + 8 * h) * 32 + t * 16 + l15] = f2bf(p);
            }
        }

        // --- ctx += P @ cols : P via per-wave LDS transpose, B from sTR ---
        v16bf pf = load_frag16(myP, 0, 32, 0, lane);
#pragma unroll
        for (int nt = 0; nt < 8; ++nt) {
            v16bf b2 = load_frag16(sTR[bufc], nt * 16, 32, 0, lane);
            ctx[nt] = wmma_bf16(pf, b2, ctx[nt]);
        }

        if (more) {
#if USE_TDM
            stage_wait();             // DMA landed; barrier publishes buffer
#else
            __syncthreads();          // readers of buf bufc^1 done
            stage_store(bufc ^ 1);
            __syncthreads();          // staged tile visible
#endif
        }
    }

    // row-sum reduction across the 16 lanes of each half-wave
#pragma unroll
    for (int r = 0; r < 8; ++r) {
        float v = dsum[r];
        v += __shfl_xor(v, 1);
        v += __shfl_xor(v, 2);
        v += __shfl_xor(v, 4);
        v += __shfl_xor(v, 8);
        dsum[r] = (v > 0.0f) ? (1.0f / v) : 0.0f;
    }

    // out += ctx / denom (this wave exclusively owns these 16 rows)
#pragma unroll
    for (int nt = 0; nt < 8; ++nt)
#pragma unroll
        for (int r = 0; r < 8; ++r) {
            size_t o = (size_t)(rb + r + 8 * h) * 128 + nt * 16 + l15;
            out[o] += ctx[nt][r] * dsum[r];
        }
}

// ---------------------------------------------------------------------------
// launch
// ---------------------------------------------------------------------------
extern "C" void kernel_launch(void* const* d_in, const int* in_sizes, int n_in,
                              void* d_out, int out_size, void* d_ws, size_t ws_size,
                              hipStream_t stream) {
    const float* user = (const float*)d_in[0];
    const float* item = (const float*)d_in[1];
    const int*   adj  = (const int*)d_in[2];
    const float* W    = (const float*)d_in[3];
    const float* bias = (const float*)d_in[4];

    const int D = in_sizes[4];            // 128
    const int N = in_sizes[0] / D;        // 16384
    const int M = in_sizes[1] / D;        // 8192

    u16* ws      = (u16*)d_ws;
    u16* w_bf    = ws;
    u16* user_bf = w_bf    + (size_t)D * D;
    u16* item_bf = user_bf + (size_t)N * D;
    u16* userT   = item_bf + (size_t)M * D;
    u16* itemT   = userT   + (size_t)N * D;

    float* out_user = (float*)d_out;
    float* out_item = out_user + (size_t)N * D;

    {   // 0) fp32 -> bf16 (+ transposed copies)
        size_t total = (size_t)(N + M + D) * D;
        int blocks = (int)((total + 255) / 256);
        convert_kernel<<<blocks, 256, 0, stream>>>(user, item, W, user_bf, item_bf,
                                                   userT, itemT, w_bf, N, M, D);
    }
    // 1) transformed = relu(x @ W^T + b) written straight into d_out
    transform_kernel<<<(N / 16) / 4, 128, 0, stream>>>(user_bf, w_bf, bias, out_user, N, D);
    transform_kernel<<<(M / 16) / 4, 128, 0, stream>>>(item_bf, w_bf, bias, out_item, M, D);

    // 2) fused masked attention, both directions (out += softmax(mask(S)) @ V)
    attn_kernel<true ><<<N / 64, 128, 0, stream>>>(user_bf, item_bf, itemT, adj,
                                                   out_user, N, M);
    attn_kernel<false><<<M / 64, 128, 0, stream>>>(item_bf, user_bf, userT, adj,
                                                   out_item, M, N);
}
